// DeformableTransformer_65446711657110
// MI455X (gfx1250) — matrix operand order repaired and checked by reading, not compiled, not run on hardware
//
#include <hip/hip_runtime.h>
#include <hip/hip_bf16.h>

// ---------------------------------------------------------------------------
// Problem constants (fixed by the reference)
// ---------------------------------------------------------------------------
#define BS_   8
#define NQ_   300
#define D_    256
#define H_    8
#define L_    4
#define P_    4
#define NL_   6
#define DFF_  1024
#define DH_   32
#define S_    13294     // 100*100 + 50*50 + 25*25 + 13*13
#define KPAD_ 320       // NQ padded up to multiple of 32 for the A@V wmma

typedef __attribute__((ext_vector_type(16))) __bf16 v16bf;
typedef __attribute__((ext_vector_type(8)))  __bf16 v8bf;
typedef __attribute__((ext_vector_type(4)))  __bf16 v4bf;
typedef __attribute__((ext_vector_type(8)))  float  v8f;

static __device__ inline int imin(int a, int b) { return a < b ? a : b; }

// native RNE fp32 -> bf16 (lowers to v_cvt_pk_bf16_f32)
static __device__ inline __bf16 f2bf(float f) { return (__bf16)f; }

// ---------------------------------------------------------------------------
// Generic GEMM:  Y[M,N] = A[M,K] (fp32->bf16) @ W[K,N] (fp32->bf16) + bias
// fp32 accumulate via v_wmma_f32_16x16x32_bf16.  K, N compile-time so the
// fully-unrolled pipeline uses immediate global offsets; M runtime.
// Block = 256 threads = 8 waves, output tile 64x64; wave (wm,wn) owns rows
// wm*16..+15 and cols wn*32..+31.
//
// LDS is double-buffered and fragment-contiguous (one aligned 32B chunk per
// lane per fragment):
//   As[buf][strip 0..3][lane 0..31][j 0..15]       (A fragments, k = 16*(lane/16) + j)
//   Wt[buf][col 0..63][k 0..31] stride 40 halfs    (transposed W)
// Out-of-range A rows are CLAMPED (not zeroed): A row r only feeds D row r,
// and out-of-range D rows are never stored, so no guard is needed in-loop.
// ---------------------------------------------------------------------------
#define WT_STRIDE 40   // halfs per column (32 + 8 pad; keeps 16B alignment)

template <int K, int N>
__global__ __launch_bounds__(256) void gemm_bf16_wmma(
    const float* __restrict__ A, const float* __restrict__ W,
    const float* __restrict__ bias, float* __restrict__ Y,
    int M, int relu)
{
    __shared__ __bf16 As[2][4 * 32 * 16];            // 2 x 4 KB
    __shared__ __bf16 Wt[2][64 * WT_STRIDE];         // 2 x 5 KB

    const int t    = threadIdx.x;
    const int wave = t >> 5;
    const int lane = t & 31;
    const int lm   = lane & 15;
    const int lhi  = lane >> 4;

    const int m0 = (int)blockIdx.y * 64;
    const int n0 = (int)blockIdx.x * 64;
    const int wm = wave & 3;             // 16-row strip
    const int wn = wave >> 2;            // 32-col strip

    // loop-invariant fragment offsets (contiguous, 16B aligned)
    const int aoff  = (wm * 32 + lane) * 16;
    const int b0off = (wn * 32 + lm) * WT_STRIDE + lhi * 16;
    const int b1off = (wn * 32 + 16 + lm) * WT_STRIDE + lhi * 16;

    // staging coordinates
    const int a_r  = t >> 3;             // 0..31 (row block; +32 on 2nd half)
    const int a_c  = (t & 7) * 4;        // 0,4,...,28 (K within chunk)
    const int a_lh = a_c >> 4;
    const int a_j  = a_c & 15;
    const int w_c  = t & 63;             // column 0..63
    const int w_rb = (t >> 6) * 8;       // k base 0,8,16,24

    // clamped (branch-free) A row pointers, hoisted out of the K loop
    const float* arow0 = A + (size_t)imin(m0 + a_r,      M - 1) * K + a_c;
    const float* arow1 = A + (size_t)imin(m0 + a_r + 32, M - 1) * K + a_c;
    const int a_dst0 = ((a_r >> 4) * 32 + a_lh * 16 + (a_r & 15)) * 16 + a_j;
    const int a_dst1 = (((a_r + 32) >> 4) * 32 + a_lh * 16 + (a_r & 15)) * 16 + a_j;
    const float* wcol = W + (size_t)w_rb * N + n0 + w_c;
    const int   wdst  = w_c * WT_STRIDE + w_rb;

    v8f acc0 = {0.f,0.f,0.f,0.f,0.f,0.f,0.f,0.f};
    v8f acc1 = acc0;

    auto stage = [&](int kn, int buf) {
        float4 v0 = *(const float4*)(arow0 + kn);
        float4 v1 = *(const float4*)(arow1 + kn);
        v4bf p0; p0[0]=f2bf(v0.x); p0[1]=f2bf(v0.y); p0[2]=f2bf(v0.z); p0[3]=f2bf(v0.w);
        v4bf p1; p1[0]=f2bf(v1.x); p1[1]=f2bf(v1.y); p1[2]=f2bf(v1.z); p1[3]=f2bf(v1.w);
        *(v4bf*)(&As[buf][a_dst0]) = p0;
        *(v4bf*)(&As[buf][a_dst1]) = p1;
        v8bf pw;
        #pragma unroll
        for (int i = 0; i < 8; ++i) pw[i] = f2bf(wcol[(size_t)(kn + i) * N]);
        *(v8bf*)(&Wt[buf][wdst]) = pw;
    };
    auto compute = [&](int buf) {
        const v16bf af  = *(const v16bf*)(&As[buf][aoff]);
        const v16bf bf0 = *(const v16bf*)(&Wt[buf][b0off]);
        const v16bf bf1 = *(const v16bf*)(&Wt[buf][b1off]);
        acc0 = __builtin_amdgcn_wmma_f32_16x16x32_bf16(false, af, false, bf0,
                                                       (short)0, acc0, false, false);
        acc1 = __builtin_amdgcn_wmma_f32_16x16x32_bf16(false, af, false, bf1,
                                                       (short)0, acc1, false, false);
    };

    stage(0, 0);
    __syncthreads();
    int cur = 0;
    #pragma unroll
    for (int k0 = 0; k0 < K - 32; k0 += 32) {
        stage(k0 + 32, cur ^ 1);   // overlaps with the wmma below
        compute(cur);
        __syncthreads();
        cur ^= 1;
    }
    compute(cur);                  // peeled last chunk: no stage, no barrier

    // Epilogue: base pointer + compile-time row offsets (r*N immediates).
    // C/D layout: lane col = lm, rows = rbase + r.
    const int rbase = m0 + wm * 16 + lhi * 8;
    const int rmax  = M - rbase;                 // rows remaining (may be <8)
    const int gcol0 = n0 + wn * 32 + lm;
    const float b0 = bias ? bias[gcol0] : 0.f;
    const float b1 = bias ? bias[gcol0 + 16] : 0.f;
    float* yp = Y + (size_t)rbase * N + gcol0;
    #pragma unroll
    for (int r = 0; r < 8; ++r) {
        if (r < rmax) {
            float y0 = acc0[r] + b0;
            float y1 = acc1[r] + b1;
            if (relu) { y0 = fmaxf(y0, 0.f); y1 = fmaxf(y1, 0.f); }
            yp[(size_t)r * N]      = y0;
            yp[(size_t)r * N + 16] = y1;
        }
    }
}

// ---------------------------------------------------------------------------
// Self-attention scores: S[b,h,q,j] = dot(Q[b,q,h,:], K[b,j,h,:]) / sqrt(DH)
// One wave per 16x16 tile; DH=32 -> exactly one wmma per tile.
// ---------------------------------------------------------------------------
__global__ __launch_bounds__(32) void attn_scores_wmma(
    const float* __restrict__ Q, const float* __restrict__ Kb,
    float* __restrict__ Sc, float inv_scale)
{
    const int bh = blockIdx.z, b = bh / H_, h = bh % H_;
    const int m0 = (int)blockIdx.y * 16, n0 = (int)blockIdx.x * 16;
    const int lane = threadIdx.x, lm = lane & 15, lhi = lane >> 4;

    v16bf af, bfm;
    const int qrow = imin(m0 + lm, NQ_ - 1);
    const float* qp = Q + ((size_t)b * NQ_ + qrow) * D_ + h * DH_ + lhi * 8;
    {
        float4 v0 = *(const float4*)(qp + 0);
        float4 v1 = *(const float4*)(qp + 4);
        float4 v2 = *(const float4*)(qp + 16);
        float4 v3 = *(const float4*)(qp + 20);
        af[0]=f2bf(v0.x); af[1]=f2bf(v0.y); af[2]=f2bf(v0.z); af[3]=f2bf(v0.w);
        af[4]=f2bf(v1.x); af[5]=f2bf(v1.y); af[6]=f2bf(v1.z); af[7]=f2bf(v1.w);
        af[8]=f2bf(v2.x); af[9]=f2bf(v2.y); af[10]=f2bf(v2.z); af[11]=f2bf(v2.w);
        af[12]=f2bf(v3.x); af[13]=f2bf(v3.y); af[14]=f2bf(v3.z); af[15]=f2bf(v3.w);
    }
    const int kcol = imin(n0 + lm, NQ_ - 1);
    const float* kp = Kb + ((size_t)b * NQ_ + kcol) * D_ + h * DH_ + lhi * 8;
    {
        float4 v0 = *(const float4*)(kp + 0);
        float4 v1 = *(const float4*)(kp + 4);
        float4 v2 = *(const float4*)(kp + 16);
        float4 v3 = *(const float4*)(kp + 20);
        bfm[0]=f2bf(v0.x); bfm[1]=f2bf(v0.y); bfm[2]=f2bf(v0.z); bfm[3]=f2bf(v0.w);
        bfm[4]=f2bf(v1.x); bfm[5]=f2bf(v1.y); bfm[6]=f2bf(v1.z); bfm[7]=f2bf(v1.w);
        bfm[8]=f2bf(v2.x); bfm[9]=f2bf(v2.y); bfm[10]=f2bf(v2.z); bfm[11]=f2bf(v2.w);
        bfm[12]=f2bf(v3.x); bfm[13]=f2bf(v3.y); bfm[14]=f2bf(v3.z); bfm[15]=f2bf(v3.w);
    }

    v8f acc = {0.f,0.f,0.f,0.f,0.f,0.f,0.f,0.f};
    acc = __builtin_amdgcn_wmma_f32_16x16x32_bf16(false, af, false, bfm,
                                                  (short)0, acc, false, false);

    // Epilogue: base pointer + r*NQ_ immediates; row/col guards hoisted.
    const int rbase = m0 + lhi * 8;
    const int rmax  = NQ_ - rbase;
    const bool cok  = (n0 + lm) < NQ_;
    float* sp = Sc + ((((size_t)b * H_ + h) * NQ_ + rbase)) * NQ_ + n0 + lm;
    #pragma unroll
    for (int r = 0; r < 8; ++r)
        if (cok && r < rmax) sp[(size_t)r * NQ_] = acc[r] * inv_scale;
}

// ---------------------------------------------------------------------------
// Row softmax over NQ, writing zero-padded rows of width KPAD_.
// ---------------------------------------------------------------------------
__global__ __launch_bounds__(32) void softmax_rows(
    const float* __restrict__ Sc, float* __restrict__ Pout)
{
    const size_t row = blockIdx.x;
    const float* s = Sc + row * NQ_;
    float* p = Pout + row * KPAD_;
    const int lane = threadIdx.x;

    float mx = -1e30f;
    for (int j = lane; j < NQ_; j += 32) mx = fmaxf(mx, s[j]);
    for (int o = 16; o > 0; o >>= 1) mx = fmaxf(mx, __shfl_xor(mx, o, 32));

    float sum = 0.f;
    for (int j = lane; j < NQ_; j += 32) { float e = __expf(s[j] - mx); p[j] = e; sum += e; }
    for (int o = 16; o > 0; o >>= 1) sum += __shfl_xor(sum, o, 32);

    const float inv = 1.f / sum;
    for (int j = lane; j < NQ_; j += 32) p[j] *= inv;
    for (int j = NQ_ + lane; j < KPAD_; j += 32) p[j] = 0.f;
}

// ---------------------------------------------------------------------------
// O[b,q,h*32+d] = sum_j P[b,h,q,j] * V[b,j,h*32+d]   (K padded to 320)
// ---------------------------------------------------------------------------
__global__ __launch_bounds__(32) void attn_av_wmma(
    const float* __restrict__ Pp, const float* __restrict__ V,
    float* __restrict__ O)
{
    const int bh = blockIdx.z, b = bh / H_, h = bh % H_;
    const int m0 = (int)blockIdx.y * 16, n0 = (int)blockIdx.x * 16;   // n0 in {0,16}
    const int lane = threadIdx.x, lm = lane & 15, lhi = lane >> 4;

    v8f acc = {0.f,0.f,0.f,0.f,0.f,0.f,0.f,0.f};
    const float* prow = Pp + (((size_t)b * H_ + h) * NQ_ + imin(m0 + lm, NQ_ - 1)) * KPAD_;
    const float* vcol = V + (size_t)b * NQ_ * D_ + h * DH_ + n0 + lm;

    auto loadA = [&](int k0, v16bf& af) {
        const float* ap = prow + k0 + lhi * 8;
        float4 v0 = *(const float4*)(ap + 0);
        float4 v1 = *(const float4*)(ap + 4);
        float4 v2 = *(const float4*)(ap + 16);
        float4 v3 = *(const float4*)(ap + 20);
        af[0]=f2bf(v0.x); af[1]=f2bf(v0.y); af[2]=f2bf(v0.z); af[3]=f2bf(v0.w);
        af[4]=f2bf(v1.x); af[5]=f2bf(v1.y); af[6]=f2bf(v1.z); af[7]=f2bf(v1.w);
        af[8]=f2bf(v2.x); af[9]=f2bf(v2.y); af[10]=f2bf(v2.z); af[11]=f2bf(v2.w);
        af[12]=f2bf(v3.x); af[13]=f2bf(v3.y); af[14]=f2bf(v3.z); af[15]=f2bf(v3.w);
    };

    // k0 = 0..256: all V row indices < 300, no clamping needed
    #pragma unroll
    for (int k0 = 0; k0 < 288; k0 += 32) {
        v16bf af, bfm;
        loadA(k0, af);
        #pragma unroll
        for (int j = 0; j < 16; ++j)
            bfm[j] = f2bf(vcol[(size_t)(k0 + lhi * 8 + (j & 7) + ((j >> 3) << 4)) * D_]);
        acc = __builtin_amdgcn_wmma_f32_16x16x32_bf16(false, af, false, bfm,
                                                      (short)0, acc, false, false);
    }
    // final chunk k0 = 288: indices 288..319, clamp (probs are 0 past NQ)
    {
        v16bf af, bfm;
        loadA(288, af);
        #pragma unroll
        for (int j = 0; j < 16; ++j) {
            int kk = imin(288 + lhi * 8 + (j & 7) + ((j >> 3) << 4), NQ_ - 1);
            bfm[j] = f2bf(vcol[(size_t)kk * D_]);
        }
        acc = __builtin_amdgcn_wmma_f32_16x16x32_bf16(false, af, false, bfm,
                                                      (short)0, acc, false, false);
    }

    // Epilogue: base pointer + r*D_ immediates.
    const int rbase = m0 + lhi * 8;
    const int rmax  = NQ_ - rbase;
    float* op = O + ((size_t)b * NQ_ + rbase) * D_ + h * DH_ + n0 + lm;
    #pragma unroll
    for (int r = 0; r < 8; ++r)
        if (r < rmax) op[(size_t)r * D_] = acc[r];
}

// ---------------------------------------------------------------------------
// Elementwise add
// ---------------------------------------------------------------------------
__global__ __launch_bounds__(256) void add_vec(
    const float* __restrict__ a, const float* __restrict__ b,
    float* __restrict__ o, int n)
{
    int i = (int)blockIdx.x * 256 + threadIdx.x;
    if (i < n) o[i] = a[i] + b[i];
}

// ---------------------------------------------------------------------------
// Out = LayerNorm(X + Yadd) * g + beta.  One block (256 thr) per row, D=256.
// Safe for Out == X (all reads happen before the first barrier).
// ---------------------------------------------------------------------------
__global__ __launch_bounds__(256) void add_layernorm(
    const float* X, const float* Yadd,
    const float* __restrict__ g, const float* __restrict__ beta, float* Out)
{
    const size_t row = blockIdx.x;
    const int t = threadIdx.x;
    __shared__ float red[256];

    float v = X[row * D_ + t] + Yadd[row * D_ + t];
    red[t] = v; __syncthreads();
    for (int s = 128; s > 0; s >>= 1) { if (t < s) red[t] += red[t + s]; __syncthreads(); }
    const float mean = red[0] * (1.f / D_);
    __syncthreads();

    const float d = v - mean;
    red[t] = d * d; __syncthreads();
    for (int s = 128; s > 0; s >>= 1) { if (t < s) red[t] += red[t + s]; __syncthreads(); }
    const float var = red[0] * (1.f / D_);

    Out[row * D_ + t] = d * rsqrtf(var + 1e-5f) * g[t] + beta[t];
}

// ---------------------------------------------------------------------------
// Deformable sampling: one block per (b,q), 256 threads = (head h, channel d).
// ---------------------------------------------------------------------------
__global__ __launch_bounds__(256) void deform_sample(
    const float* __restrict__ val,    // [BS,S,D]
    const float* __restrict__ off,    // [BS,NQ,H,L,P,2]
    const float* __restrict__ awl,    // [BS,NQ,H,16] logits
    const float* __restrict__ refp,   // [BS,NQ,2]
    const float* __restrict__ vr,     // [BS,L,2]
    float* __restrict__ out)          // [BS,NQ,D]
{
    const int shp[4]    = {100, 50, 25, 13};
    const int lstart[4] = {0, 10000, 12500, 13125};

    const int bq = blockIdx.x;
    const int b = bq / NQ_;
    const int t = threadIdx.x;
    const int h = t >> 5;
    const int d = t & 31;

    const float* lg = awl + ((size_t)bq * H_ + h) * 16;
    float mx = lg[0];
    #pragma unroll
    for (int i = 1; i < 16; ++i) mx = fmaxf(mx, lg[i]);
    float w[16]; float sum = 0.f;
    #pragma unroll
    for (int i = 0; i < 16; ++i) { w[i] = __expf(lg[i] - mx); sum += w[i]; }
    const float inv = 1.f / sum;

    const float rx = refp[(size_t)bq * 2 + 0];
    const float ry = refp[(size_t)bq * 2 + 1];
    const float* ofp = off + ((size_t)bq * H_ + h) * (L_ * P_ * 2);

    float accum = 0.f;
    for (int l = 0; l < L_; ++l) {
        const int Hl = shp[l], Wl = shp[l], ls = lstart[l];
        const float refx = rx * vr[((size_t)b * L_ + l) * 2 + 0];
        const float refy = ry * vr[((size_t)b * L_ + l) * 2 + 1];
        const float* vbase = val + ((size_t)b * S_ + ls) * D_ + h * DH_ + d;
        for (int p = 0; p < P_; ++p) {
            const float locx = refx + ofp[(l * P_ + p) * 2 + 0] / (float)Wl;
            const float locy = refy + ofp[(l * P_ + p) * 2 + 1] / (float)Hl;
            const float x = locx * Wl - 0.5f;
            const float y = locy * Hl - 0.5f;
            const float x0f = floorf(x), y0f = floorf(y);
            const int x0 = (int)x0f, y0 = (int)y0f;
            const float lx = x - x0f, ly = y - y0f;
            const float wgt = w[l * P_ + p] * inv;

            float sv = 0.f;
            #pragma unroll
            for (int cy = 0; cy < 2; ++cy) {
                #pragma unroll
                for (int cx = 0; cx < 2; ++cx) {
                    const int xi = x0 + cx, yi = y0 + cy;
                    if (xi >= 0 && xi < Wl && yi >= 0 && yi < Hl) {
                        const float cw = (cx ? lx : 1.f - lx) * (cy ? ly : 1.f - ly);
                        sv += cw * vbase[((size_t)yi * Wl + xi) * D_];
                    }
                }
            }
            accum += wgt * sv;
        }
    }
    out[(size_t)bq * D_ + h * DH_ + d] = accum;
}

// ---------------------------------------------------------------------------
// Host orchestration
// ---------------------------------------------------------------------------
extern "C" void kernel_launch(void* const* d_in, const int* in_sizes, int n_in,
                              void* d_out, int out_size, void* d_ws, size_t ws_size,
                              hipStream_t stream)
{
    (void)in_sizes; (void)n_in; (void)ws_size;

    const float* tgt    = (const float*)d_in[0];
    const float* qpos   = (const float*)d_in[1];
    const float* refp   = (const float*)d_in[2];
    const float* src    = (const float*)d_in[3];
    /* d_in[4] = src_spatial_shapes (compile-time constants used) */
    const float* vr     = (const float*)d_in[5];
    const float* w_samp = (const float*)d_in[6];   const float* b_samp = (const float*)d_in[7];
    const float* w_attw = (const float*)d_in[8];   const float* b_attw = (const float*)d_in[9];
    const float* w_val  = (const float*)d_in[10];  const float* b_val  = (const float*)d_in[11];
    const float* w_cout = (const float*)d_in[12];  const float* b_cout = (const float*)d_in[13];
    const float* w_q    = (const float*)d_in[14];  const float* b_q    = (const float*)d_in[15];
    const float* w_k    = (const float*)d_in[16];  const float* b_k    = (const float*)d_in[17];
    const float* w_v    = (const float*)d_in[18];  const float* b_v    = (const float*)d_in[19];
    const float* w_so   = (const float*)d_in[20];  const float* b_so   = (const float*)d_in[21];
    const float* w_ff1  = (const float*)d_in[22];  const float* b_ff1  = (const float*)d_in[23];
    const float* w_ff2  = (const float*)d_in[24];  const float* b_ff2  = (const float*)d_in[25];
    const float* ln1g   = (const float*)d_in[26];  const float* ln1b   = (const float*)d_in[27];
    const float* ln2g   = (const float*)d_in[28];  const float* ln2b   = (const float*)d_in[29];
    const float* ln3g   = (const float*)d_in[30];  const float* ln3b   = (const float*)d_in[31];

    char* ws = (char*)d_ws;
    size_t woff = 0;
    auto alloc = [&](size_t nfloats) -> float* {
        float* p = (float*)(ws + woff);
        woff += ((nfloats * sizeof(float) + 255) & ~(size_t)255);
        return p;
    };

    const size_t NA = (size_t)BS_ * NQ_ * D_;         // 614400
    float* outb   = alloc(NA);
    float* qk     = alloc(NA);
    float* qb     = alloc(NA);
    float* kb     = alloc(NA);
    float* vb     = alloc(NA);
    float* tmp    = alloc(NA);
    float* ao     = alloc(NA);
    float* scores = alloc((size_t)BS_ * H_ * NQ_ * NQ_);
    float* probs  = alloc((size_t)BS_ * H_ * NQ_ * KPAD_);
    float* offb   = alloc(NA);                        // H*L*P*2 == 256 == D
    float* awlb   = alloc((size_t)BS_ * NQ_ * H_ * 16);
    float* ffh    = alloc((size_t)BS_ * NQ_ * DFF_);
    float* valb   = alloc((size_t)BS_ * S_ * D_);

    const int M   = BS_ * NQ_;                        // 2400
    const int MT  = (M + 63) / 64;                    // 38
    const int MV  = BS_ * S_;                         // 106352
    const int MVT = (MV + 63) / 64;                   // 1662
    const int NT  = (NQ_ + 15) / 16;                  // 19
    const float inv_scale = 0.17677669529663688f;     // 1/sqrt(32)

    hipMemcpyAsync(outb, tgt, NA * sizeof(float), hipMemcpyDeviceToDevice, stream);

    for (int l = 0; l < NL_; ++l) {
        const size_t dd = (size_t)l * D_ * D_;
        // --- self attention -------------------------------------------------
        add_vec<<<(int)((NA + 255) / 256), 256, 0, stream>>>(outb, qpos, qk, (int)NA);
        gemm_bf16_wmma<256, 256><<<dim3(D_ / 64, MT), 256, 0, stream>>>(qk,   w_q + dd, b_q + (size_t)l * D_, qb, M, 0);
        gemm_bf16_wmma<256, 256><<<dim3(D_ / 64, MT), 256, 0, stream>>>(qk,   w_k + dd, b_k + (size_t)l * D_, kb, M, 0);
        gemm_bf16_wmma<256, 256><<<dim3(D_ / 64, MT), 256, 0, stream>>>(outb, w_v + dd, b_v + (size_t)l * D_, vb, M, 0);
        attn_scores_wmma<<<dim3(NT, NT, BS_ * H_), 32, 0, stream>>>(qb, kb, scores, inv_scale);
        softmax_rows<<<BS_ * H_ * NQ_, 32, 0, stream>>>(scores, probs);
        attn_av_wmma<<<dim3(2, NT, BS_ * H_), 32, 0, stream>>>(probs, vb, ao);
        gemm_bf16_wmma<256, 256><<<dim3(D_ / 64, MT), 256, 0, stream>>>(ao, w_so + dd, b_so + (size_t)l * D_, tmp, M, 0);
        add_layernorm<<<M, 256, 0, stream>>>(outb, tmp, ln2g + (size_t)l * D_, ln2b + (size_t)l * D_, outb);

        // --- deformable attention ------------------------------------------
        add_vec<<<(int)((NA + 255) / 256), 256, 0, stream>>>(outb, qpos, qk, (int)NA);
        gemm_bf16_wmma<256, 256><<<dim3(D_ / 64, MVT), 256, 0, stream>>>(src, w_val + dd, b_val + (size_t)l * D_, valb, MV, 0);
        gemm_bf16_wmma<256, 256><<<dim3(256 / 64, MT), 256, 0, stream>>>(qk, w_samp + (size_t)l * D_ * 256, b_samp + (size_t)l * 256, offb, M, 0);
        gemm_bf16_wmma<256, 128><<<dim3(128 / 64, MT), 256, 0, stream>>>(qk, w_attw + (size_t)l * D_ * 128, b_attw + (size_t)l * 128, awlb, M, 0);
        deform_sample<<<M, 256, 0, stream>>>(valb, offb, awlb, refp, vr, ao);
        gemm_bf16_wmma<256, 256><<<dim3(D_ / 64, MT), 256, 0, stream>>>(ao, w_cout + dd, b_cout + (size_t)l * D_, tmp, M, 0);
        add_layernorm<<<M, 256, 0, stream>>>(outb, tmp, ln1g + (size_t)l * D_, ln1b + (size_t)l * D_, outb);

        // --- FFN ------------------------------------------------------------
        gemm_bf16_wmma<256, 1024><<<dim3(DFF_ / 64, MT), 256, 0, stream>>>(outb, w_ff1 + (size_t)l * D_ * DFF_, b_ff1 + (size_t)l * DFF_, ffh, M, 1);
        gemm_bf16_wmma<1024, 256><<<dim3(D_ / 64, MT), 256, 0, stream>>>(ffh, w_ff2 + (size_t)l * DFF_ * D_, b_ff2 + (size_t)l * D_, tmp, M, 0);
        add_layernorm<<<M, 256, 0, stream>>>(outb, tmp, ln3g + (size_t)l * D_, ln3b + (size_t)l * D_, outb);
    }

    hipMemcpyAsync(d_out, outb, (size_t)out_size * sizeof(float),
                   hipMemcpyDeviceToDevice, stream);
}
